// moe_stochastic_model_39582418600227
// MI455X (gfx1250) — matrix operand order repaired and checked by reading, not compiled, not run on hardware
//
#include <hip/hip_runtime.h>
#include <hip/hip_bf16.h>

// MoE stochastic model for MI455X (gfx1250, wave32, WMMA + TDM).
//  - GEMM2 (per expert: [16x128]x[128x16]) -> v_wmma_f32_16x16x32_f16 (4 per expert)
//  - GEMM1 (per expert: [16x4 ]x[4 x128]) -> v_wmma_f32_16x16x4_f32  (8 per expert)
//  - w1/b1 staged to LDS by the Tensor Data Mover (tensor_load_to_lds, TENSORcnt)
//  - attention branch is provably dead (softmax over singleton dim) => context = sum_e y
//  - sampling reproduces jax threefry2x32(key=[0,42]) gumbel-max exactly

typedef __attribute__((ext_vector_type(16))) _Float16 v16h;
typedef __attribute__((ext_vector_type(8)))  float    v8f;
typedef __attribute__((ext_vector_type(4)))  float    v4f;
typedef __attribute__((ext_vector_type(2)))  float    v2f;
typedef __attribute__((ext_vector_type(4)))  unsigned u32x4;
typedef __attribute__((ext_vector_type(8)))  unsigned u32x8;

#define NROWS   500000
#define NTILES  31250     // 500000 / 16
#define NBLK    1953
#define WPB     4         // waves per block (128 threads)
#define NE      16
#define NH      128
#define NC      10
#define NGATE   12

__device__ __forceinline__ unsigned rotl32(unsigned x, unsigned r) {
  return (x << r) | (x >> (32u - r));
}

// JAX threefry2x32 with key = [0, 42] (jax.random.key(42))
__device__ __forceinline__ unsigned threefry42(unsigned x0, unsigned x1, int which) {
  const unsigned ks0 = 0u, ks1 = 42u;
  const unsigned ks2 = 0x1BD11BDAu ^ ks0 ^ ks1;
  const unsigned ks[3] = {ks0, ks1, ks2};
  const unsigned R0[4] = {13u, 15u, 26u, 6u};
  const unsigned R1[4] = {17u, 29u, 16u, 24u};
  x0 += ks[0];
  x1 += ks[1];
#pragma unroll
  for (int i = 0; i < 5; ++i) {
#pragma unroll
    for (int j = 0; j < 4; ++j) {
      unsigned r = (i & 1) ? R1[j] : R0[j];
      x0 += x1;
      x1 = rotl32(x1, r);
      x1 ^= x0;
    }
    x0 += ks[(i + 1) % 3];
    x1 += ks[(i + 2) % 3] + (unsigned)(i + 1);
  }
  return which ? x1 : x0;
}

// LDS byte offset of a __shared__ object: per CDNA5 aperture rule the low 32
// bits of the flat address ARE the LDS address.
__device__ __forceinline__ unsigned lds_off(const void* p) {
  return (unsigned)(unsigned long long)p;
}

// Issue a 1-D TDM copy of n f32 elements from global to LDS.
// D# group0: count=1 | lds_addr[63:32] | global_addr[120:64] | type=2[127:126]
// D# group1: data_size=2 (4B), tensor_dim0=n, tensor_dim1=1, tile_dim0=n,
//            tile_dim1=1, tensor_dim0_stride=n  (n must fit in 16 bits)
__device__ __forceinline__ void tdm_load_f32(unsigned ldsaddr, const float* gptr, unsigned n) {
  unsigned long long ga = (unsigned long long)gptr;
  u32x4 g0;
  g0[0] = 1u;                                   // count=1, user descriptor
  g0[1] = ldsaddr;                              // lds_addr
  g0[2] = (unsigned)(ga & 0xFFFFFFFFu);         // global_addr[31:0]
  g0[3] = (unsigned)((ga >> 32) & 0x01FFFFFFu)  // global_addr[56:32]
        | (2u << 30);                           // type = 2 ("image")
  u32x8 g1;
  g1[0] = 0x00020000u;                          // data_size = 2 -> 4 bytes
  g1[1] = (n & 0xFFFFu) << 16;                  // tensor_dim0[15:0] @ bits 63:48
  g1[2] = ((n >> 16) & 0xFFFFu) | (1u << 16);   // tensor_dim0[31:16]; tensor_dim1 = 1
  g1[3] = (n & 0xFFFFu) << 16;                  // tile_dim0 @ bits 127:112
  g1[4] = 1u;                                   // tile_dim1 = 1, tile_dim2 = 0
  g1[5] = n;                                    // tensor_dim0_stride[31:0]
  g1[6] = 0u;                                   // stride hi, dim1 stride lo
  g1[7] = 0u;
  asm volatile("tensor_load_to_lds %0, %1" :: "s"(g0), "s"(g1) : "memory");
}

__global__ __launch_bounds__(128)
void moe_stoch_kernel(const float* __restrict__ inp,   // [B,2]
                      const float* __restrict__ w1,    // [E,2,H]
                      const float* __restrict__ b1,    // [E,H]
                      const float* __restrict__ w2,    // [E,H,C]
                      const float* __restrict__ b2,    // [E,C]
                      const float* __restrict__ Wg,    // [E,12]
                      const float* __restrict__ bg,    // [E]
                      float* __restrict__ out)         // [B,C]
{
  // --- LDS ---
  __shared__ __align__(16) _Float16 w2frag[NE * 4 * 32 * 16]; // 64 KB: f16 B-fragments, all experts
  __shared__ __align__(16) float    w1s[NE * 2 * NH];         // 16 KB: TDM-staged w1
  __shared__ __align__(16) float    b1s[NE * NH];             //  8 KB: TDM-staged b1
  __shared__ float    b2p[NE * 16];                           // bias padded to 16 cols
  __shared__ float    wgW[NE * NGATE];
  __shared__ float    wgB[NE];
  __shared__ __align__(16) _Float16 h16[WPB][16][NH];         // per-wave h tile (f16)
  __shared__ float    ytile[WPB][16][NE][NC];                 // per-wave y (all experts)
  __shared__ float    ctile[WPB][16][NC];                     // per-wave context
  __shared__ float    xtile[WPB][16][2];
  __shared__ unsigned smp[WPB][16];

  const int tid = threadIdx.x;

  // ---- kick off TDM DMA of w1/b1 into LDS (overlaps the w2 conversion below) ----
  if (tid == 0) {
    tdm_load_f32(lds_off(w1s), w1, NE * 2 * NH); // 4096 floats
    tdm_load_f32(lds_off(b1s), b1, NE * NH);     // 2048 floats
  }

  // ---- stage w2 -> f16 B-fragment layout (K = (lane<16?0:16)+t within 32-K chunk) ----
  for (int idx = tid; idx < NE * 4 * 32 * 16; idx += 128) {
    int t    = idx & 15;
    int lane = (idx >> 4) & 31;
    int kc   = (idx >> 9) & 3;
    int e    = idx >> 11;
    int K = kc * 32 + ((lane < 16) ? 0 : 16) + t;
    int N = lane & 15;
    float val = (N < NC) ? w2[(e * NH + K) * NC + N] : 0.0f;
    w2frag[idx] = (_Float16)val;
  }
  for (int idx = tid; idx < NE * 16; idx += 128) {
    int n = idx & 15;
    b2p[idx] = (n < NC) ? b2[(idx >> 4) * NC + n] : 0.0f;
  }
  for (int idx = tid; idx < NE * NGATE; idx += 128) wgW[idx] = Wg[idx];
  if (tid < NE) wgB[tid] = bg[tid];

  __builtin_amdgcn_s_wait_tensorcnt(0);  // TDM completion (TENSORcnt)
  __syncthreads();

  const int wid = tid >> 5;
  const int L   = tid & 31;
  const int lo  = (L < 16);
  const int n16 = L & 15;

  for (int tile = blockIdx.x * WPB + wid; tile < NTILES; tile += NBLK * WPB) {
    const int rowbase = tile * 16;

    // ---- load the 16x2 input tile ----
    float x0 = 0.0f, x1 = 0.0f;
    if (lo) {
      v2f xv = ((const v2f*)inp)[rowbase + L];
      x0 = xv.x; x1 = xv.y;
      xtile[wid][L][0] = x0;
      xtile[wid][L][1] = x1;
    }
    // f32 A-fragment 16x4: lanes 0-15 hold K=0,1; lanes 16-31 hold K=2,3 (zero pad, D=2)
    v2f afrag;
    afrag.x = x0;
    afrag.y = x1;

    // prefetch next tile's inputs
    {
      int nt2 = tile + NBLK * WPB;
      if (nt2 < NTILES) __builtin_prefetch(&inp[(size_t)nt2 * 32], 0, 3);
    }

    v8f ctx = {};

    for (int e = 0; e < NE; ++e) {
      // ---------- GEMM1: h = relu(x @ w1[e] + b1[e]) via f32 WMMA ----------
#pragma unroll
      for (int nt = 0; nt < 8; ++nt) {
        int col = nt * 16 + n16;
        // unconditional LDS loads + cndmask select (no exec-mask branching)
        float t0 = w1s[(e * 2 + 0) * NH + col];
        float t1 = w1s[(e * 2 + 1) * NH + col];
        v2f bfrag;
        bfrag.x = lo ? t0 : 0.0f; // f32 B-fragment 4x16: lanes 16-31 are K=2,3 pad
        bfrag.y = lo ? t1 : 0.0f;
        v8f hz = {};
        hz = __builtin_amdgcn_wmma_f32_16x16x4_f32(
            false, afrag, false, bfrag, (short)0, hz, false, false);
        float bias = b1s[e * NH + col];
#pragma unroll
        for (int r = 0; r < 8; ++r) {
          int row = r + (lo ? 0 : 8); // f32 C layout: lanes 0-15 M=r, lanes 16-31 M=r+8
          float hv = fmaxf(hz[r] + bias, 0.0f);
          h16[wid][row][col] = (_Float16)hv;
        }
      }

      // ---------- GEMM2: logits = h @ w2[e] via f16 WMMA, K=128 ----------
      v8f acc = {};
#pragma unroll
      for (int kc = 0; kc < 4; ++kc) {
        // f16 A-fragment 16x32: lane M=n16; halves 0-15 at K=k0..k0+15 (k0 = 16*half)
        int k0 = kc * 32 + (lo ? 0 : 16);
        union { v16h v; v4f f[2]; } A, B;
        A.f[0] = *(const v4f*)&h16[wid][n16][k0];
        A.f[1] = *(const v4f*)&h16[wid][n16][k0 + 8];
        const _Float16* bp = &w2frag[(((e * 4 + kc) * 32) + L) * 16];
        B.f[0] = ((const v4f*)bp)[0];
        B.f[1] = ((const v4f*)bp)[1];
        acc = __builtin_amdgcn_wmma_f32_16x16x32_f16(
            false, A.v, false, B.v, (short)0, acc, false, false);
      }

      // ---------- per-row softmax over C=10 (lanes 0..9 of each 16-lane group) ----------
      const bool  valid = (n16 < NC);
      const float bias2 = b2p[e * 16 + n16];
#pragma unroll
      for (int r = 0; r < 8; ++r) {
        float lg = valid ? (acc[r] + bias2) : -3.0e38f;
        float m = lg;
        m = fmaxf(m, __shfl_xor(m, 1, 32));
        m = fmaxf(m, __shfl_xor(m, 2, 32));
        m = fmaxf(m, __shfl_xor(m, 4, 32));
        m = fmaxf(m, __shfl_xor(m, 8, 32));
        float ex = valid ? __expf(lg - m) : 0.0f;
        float s = ex;
        s += __shfl_xor(s, 1, 32);
        s += __shfl_xor(s, 2, 32);
        s += __shfl_xor(s, 4, 32);
        s += __shfl_xor(s, 8, 32);
        float yv = ex / s;
        ctx[r] += yv; // context = sum over experts (attention is all-ones)
        if (valid) ytile[wid][r + (lo ? 0 : 8)][e][n16] = yv;
      }
    }

    // ---- spill context to LDS for the gate stage ----
#pragma unroll
    for (int r = 0; r < 8; ++r)
      if (n16 < NC) ctile[wid][r + (lo ? 0 : 8)][n16] = ctx[r];

    // ---- gate logits + exact JAX gumbel-max sampling; lane = expert f ----
    float wg[NGATE];
#pragma unroll
    for (int j = 0; j < NGATE; ++j) wg[j] = wgW[n16 * NGATE + j];
    const float bgv = wgB[n16];

#pragma unroll
    for (int r = 0; r < 8; ++r) {
      int row = r + (lo ? 0 : 8);
      float gl = fmaf(xtile[wid][row][0], wg[0],
                 fmaf(xtile[wid][row][1], wg[1], bgv));
#pragma unroll
      for (int c = 0; c < NC; ++c) gl = fmaf(ctile[wid][row][c], wg[2 + c], gl);

      // gumbel noise at flat index i = b*16 + f, threefry counter split-halves
      unsigned i = (unsigned)(rowbase + row) * 16u + (unsigned)n16;
      const unsigned HALF = (unsigned)(NROWS * NE / 2); // 4,000,000
      unsigned bits = (i < HALF) ? threefry42(i, i + HALF, 0)
                                 : threefry42(i - HALF, i, 1);
      float f01 = __uint_as_float(0x3f800000u | (bits >> 9)) - 1.0f;
      const float TINY = 1.1754944e-38f;
      float u = fmaxf(f01 * (1.0f - TINY) + TINY, TINY);
      float g = -__logf(-__logf(u));
      float z = gl + g; // argmax(log p + g) == argmax(logit + g)

      float best = z;
      unsigned bi = (unsigned)n16;
#pragma unroll
      for (int mask = 1; mask <= 8; mask <<= 1) {
        float    ov = __shfl_xor(best, mask, 32);
        unsigned ob = (unsigned)__shfl_xor((int)bi, mask, 32);
        if (ov > best || (ov == best && ob < bi)) { best = ov; bi = ob; }
      }
      if (n16 == 0) smp[wid][row] = bi;
    }

    // ---- gather the sampled expert's y and store ----
    if (lo) {
      unsigned s = smp[wid][L];
      v2f* op = (v2f*)out + (size_t)(rowbase + L) * 5;
#pragma unroll
      for (int j = 0; j < 5; ++j) {
        v2f vv;
        vv.x = ytile[wid][L][s][2 * j + 0];
        vv.y = ytile[wid][L][s][2 * j + 1];
        op[j] = vv;
      }
    }
  }
}

extern "C" void kernel_launch(void* const* d_in, const int* in_sizes, int n_in,
                              void* d_out, int out_size, void* d_ws, size_t ws_size,
                              hipStream_t stream) {
  (void)in_sizes; (void)n_in; (void)out_size; (void)d_ws; (void)ws_size;
  const float* inp = (const float*)d_in[0];
  const float* w1  = (const float*)d_in[1];
  const float* b1  = (const float*)d_in[2];
  const float* w2  = (const float*)d_in[3];
  const float* b2  = (const float*)d_in[4];
  // d_in[5]=Wx, d_in[6]=We, d_in[7]=v are mathematically dead (softmax over singleton axis)
  const float* Wg  = (const float*)d_in[8];
  const float* bg  = (const float*)d_in[9];
  float* out = (float*)d_out;

  hipLaunchKernelGGL(moe_stoch_kernel, dim3(NBLK), dim3(128), 0, stream,
                     inp, w1, b1, w2, b2, Wg, bg, out);
}
